// Palm_57930518889103
// MI455X (gfx1250) — matrix operand order repaired and checked by reading, not compiled, not run on hardware
//
#include <hip/hip_runtime.h>
#include <hip/hip_bf16.h>

#define NN   100000
#define DD   512
#define KK   400
#define NCLS 100
#define INV_EPS 20.0f      // 1/0.05
#define INV_TEMP 10.0f     // 1/0.1
#define CHUNK 8192
#define NCHUNK 13          // ceil(NN/CHUNK)
#define NCB 1563           // ceil(NN/64)   sinkhorn slab blocks
#define NUB 391            // ceil(NN/256)  um colpart blocks

typedef __attribute__((ext_vector_type(16))) __bf16 v16bf;
typedef __attribute__((ext_vector_type(8)))  __bf16 v8bf;
typedef __attribute__((ext_vector_type(8)))  float  v8f;
typedef __attribute__((ext_vector_type(4)))  unsigned int v4u;
typedef __attribute__((ext_vector_type(8)))  int v8i;
typedef __attribute__((ext_vector_type(4)))  int v4i;

// ---------------- utility kernels ----------------
__global__ void conv_bf16(const float* __restrict__ src, __bf16* __restrict__ dst, long n) {
    long i = (long)blockIdx.x * blockDim.x + threadIdx.x;
    if (i < n) dst[i] = (__bf16)src[i];
}

// ---------------- NT WMMA GEMM: C[M,Kc] = A[M,512] * B[Kc,512]^T ----------------
// block = 8 waves x 16 rows (128 rows) x 64 cols; each wave: 1 A-frag -> 4 wmma.
// B panel (64x512 bf16 = 64KB) staged into LDS by the Tensor Data Mover when available.
__global__ __launch_bounds__(256)
void gemm_nt(const __bf16* __restrict__ A, const __bf16* __restrict__ B,
             float* __restrict__ C, int M, int Kc) {
    __shared__ __bf16 Bs[64][DD];   // 64 KB
    const int tid  = threadIdx.x;
    const int wave = tid >> 5, lane = tid & 31;
    const int colBase = blockIdx.y * 64;
    const int rowBase = blockIdx.x * 128 + wave * 16;

#if __has_builtin(__builtin_amdgcn_tensor_load_to_lds)
    // ---- TDM path: one 2-D tile descriptor; hardware zero-fills rows >= (Kc-colBase) ----
    if (wave == 0) {
        unsigned long long gaddr = (unsigned long long)(const void*)(B + (long)colBase * DD);
        unsigned int laddr = (unsigned int)(unsigned long long)(const void*)&Bs[0][0];
        unsigned int rem = (unsigned int)(Kc - colBase);      // remaining tensor rows
        v4u g0;
        g0[0] = 1u;                                           // count=1 (valid user D#)
        g0[1] = laddr;                                        // lds_addr
        g0[2] = (unsigned int)gaddr;                          // global_addr[31:0]
        g0[3] = (unsigned int)((gaddr >> 32) & 0x01FFFFFFu) | (2u << 30); // addr[56:32] | type=2
        v8i g1;
        g1[0] = 0x00010000;                                   // data_size=1 (2 bytes)
        g1[1] = (int)((DD & 0xFFFFu) << 16);                  // tensor_dim0 = 512 (bits 79:48)
        g1[2] = (int)((rem & 0xFFFFu) << 16);                 // tensor_dim1 = rem (bits 111:80)
        g1[3] = (int)((DD & 0xFFFFu) << 16);                  // tile_dim0 = 512 (bits 127:112)
        g1[4] = 64;                                           // tile_dim1 = 64 rows; tile_dim2=0
        g1[5] = DD;                                           // tensor_dim0_stride[31:0] = 512
        g1[6] = 0;
        g1[7] = 0;
        v4i gz4 = {0, 0, 0, 0};
        v8i gz8 = {0, 0, 0, 0, 0, 0, 0, 0};
        __builtin_amdgcn_tensor_load_to_lds(g0, g1, gz4, gz4, gz8, 0);
        __builtin_amdgcn_s_wait_tensorcnt(0);
    }
#else
    // ---- fallback: manual staging, 4096 16-byte chunks ----
    for (int cidx = tid; cidx < 4096; cidx += 256) {
        int row = cidx >> 6;
        int off = (cidx & 63) * 8;
        long brow = colBase + row; if (brow >= Kc) brow = Kc - 1;
        *(uint4*)(&Bs[row][off]) = *(const uint4*)(B + brow * DD + off);
    }
#endif
    __syncthreads();

    const int r  = lane & 15;
    const int hi = lane >> 4;
    long arow = rowBase + r; if (arow >= M) arow = M - 1;   // clamp, EXEC stays full
    const __bf16* Ap = A + arow * DD;

    v8f acc[4] = {};
#pragma unroll
    for (int kb = 0; kb < DD; kb += 32) {
        // A 16x32 bf16: lanes<16 hold K 0..7 & 16..23, lanes>=16 hold K 8..15 & 24..31
        int ka = kb + (hi ? 8 : 0);
        v8bf alo = *(const v8bf*)(Ap + ka);
        v8bf ahi = *(const v8bf*)(Ap + ka + 16);
        v16bf a;
#pragma unroll
        for (int e = 0; e < 8; ++e) { a[e] = alo[e]; a[e + 8] = ahi[e]; }
        int kbB = kb + (hi ? 8 : 0);
#pragma unroll
        for (int t = 0; t < 4; ++t) {
            // B 32x16: lane = column, lanes<16 hold K 0..7 & 16..23, lanes>=16 hold K 8..15 & 24..31
            v8bf blo = *(const v8bf*)(&Bs[t * 16 + r][kbB]);
            v8bf bhi = *(const v8bf*)(&Bs[t * 16 + r][kbB + 16]);
            v16bf b;
#pragma unroll
            for (int e = 0; e < 8; ++e) { b[e] = blo[e]; b[e + 8] = bhi[e]; }
            acc[t] = __builtin_amdgcn_wmma_f32_16x16x32_bf16(false, a, false, b,
                                                             (short)0, acc[t], false, false);
        }
    }
#pragma unroll
    for (int t = 0; t < 4; ++t) {
        int cc = colBase + t * 16 + r;
        if (cc < Kc) {
#pragma unroll
            for (int i = 0; i < 8; ++i) {
                int rr = rowBase + hi * 8 + i;
                if (rr < M) C[(long)rr * Kc + cc] = acc[t][i];
            }
        }
    }
}

// ---------------- Sinkhorn (scaling-vector form) ----------------
// iteration 1 column-partials with c == 1
__global__ __launch_bounds__(512)
void sink_colpass1(const float* __restrict__ S, float* __restrict__ colpart) {
    int k = threadIdx.x;
    if (k >= KK) return;
    long n0 = (long)blockIdx.x * 64;
    long nEnd = n0 + 64; if (nEnd > NN) nEnd = NN;
    float sum = 0.f;
    for (long n = n0; n < nEnd; ++n)
        sum += __expf(S[n * KK + k] * INV_EPS);
    colpart[(long)blockIdx.x * KK + k] = sum;
}

__global__ void sink_colreduce(const float* __restrict__ colpart, float* __restrict__ r) {
    int k = (int)(blockIdx.x * blockDim.x + threadIdx.x);
    if (k >= KK) return;
    float s = 0.f;
    for (int b = 0; b < NCB; ++b) s += colpart[(long)b * KK + k];   // fixed order
    r[k] = 1.0f / ((float)KK * fmaxf(s, 1e-12f));
}

// fused: per-row c (wave reduce) + next-iteration column partials (per-wave LDS bins)
__global__ __launch_bounds__(256)
void sink_fused(const float* __restrict__ S, const float* __restrict__ r,
                float* __restrict__ colpart) {
    __shared__ float part[8][KK];
    const int tid = threadIdx.x, wave = tid >> 5, lane = tid & 31;
    for (int i = tid; i < 8 * KK; i += 256) (&part[0][0])[i] = 0.f;
    __syncthreads();
    long n0 = (long)blockIdx.x * 64 + wave * 8;
    for (int rr = 0; rr < 8; ++rr) {
        long n = n0 + rr;
        bool valid = (n < NN);
        long nc = valid ? n : (NN - 1);
        float e[13];
        float rsum = 0.f;
#pragma unroll
        for (int j = 0; j < 13; ++j) {
            int k = lane + 32 * j;
            if (k < KK) { e[j] = __expf(S[nc * KK + k] * INV_EPS); rsum += r[k] * e[j]; }
            else e[j] = 0.f;
        }
        for (int off = 16; off; off >>= 1) rsum += __shfl_xor(rsum, off, 32);
        float cn = 1.0f / ((float)NN * fmaxf(rsum, 1e-12f));
        if (valid) {
#pragma unroll
            for (int j = 0; j < 13; ++j) {
                int k = lane + 32 * j;
                if (k < KK) part[wave][k] += e[j] * cn;   // lane-private slots, no races
            }
        }
    }
    __syncthreads();
    for (int k = tid; k < KK; k += 256) {
        float s = 0.f;
        for (int w = 0; w < 8; ++w) s += part[w][k];      // fixed order
        colpart[(long)blockIdx.x * KK + k] = s;
    }
}

// last row pass of Sinkhorn #1 fused with update_mask row-normalize (4 nonzeros/row)
__global__ __launch_bounds__(256)
void fused_w4(const float* __restrict__ S, const float* __restrict__ r,
              const int* __restrict__ labels, float* __restrict__ W4) {
    int wave = threadIdx.x >> 5, lane = threadIdx.x & 31;
    long n = (long)blockIdx.x * 8 + wave;
    if (n >= NN) return;
    float rsum = 0.f;
    for (int k = lane; k < KK; k += 32)
        rsum += r[k] * __expf(S[n * KK + k] * INV_EPS);
    for (int off = 16; off; off >>= 1) rsum += __shfl_xor(rsum, off, 32);
    if (lane == 0) {
        float cn = (float)NN * (1.0f / ((float)NN * fmaxf(rsum, 1e-12f)));  // B * c
        int l = labels[n];
        float q[4], a = 0.f;
#pragma unroll
        for (int p = 0; p < 4; ++p) {
            int k = l + NCLS * p;
            q[p] = r[k] * __expf(S[n * KK + k] * INV_EPS) * cn;
            a += q[p];
        }
        a = fmaxf(a, 1e-12f);
#pragma unroll
        for (int p = 0; p < 4; ++p) W4[n * 4 + p] = q[p] / a;
    }
}

// ---------------- update_mask column sums ----------------
__global__ __launch_bounds__(512)
void um_colpart(const float* __restrict__ W4, const int* __restrict__ labels,
                float* __restrict__ part) {
    __shared__ int   lbl[256];
    __shared__ float w4s[256][4];
    long n0 = (long)blockIdx.x * 256;
    int tid = threadIdx.x;
    int cnt = (int)((NN - n0) < 256 ? (NN - n0) : 256);
    if (tid < cnt) {
        lbl[tid] = labels[n0 + tid];
        for (int p = 0; p < 4; ++p) w4s[tid][p] = W4[(n0 + tid) * 4 + p];
    }
    __syncthreads();
    if (tid < KK) {
        int cls = tid % NCLS, p = tid / NCLS;
        float s = 0.f;
        for (int i = 0; i < cnt; ++i)
            if (lbl[i] == cls) s += w4s[i][p];
        part[(long)blockIdx.x * KK + tid] = s;
    }
}

__global__ void um_colreduce(const float* __restrict__ part, float* __restrict__ umcol) {
    int k = (int)(blockIdx.x * blockDim.x + threadIdx.x);
    if (k >= KK) return;
    float s = 0.f;
    for (int b = 0; b < NUB; ++b) s += part[(long)b * KK + k];
    umcol[k] = fmaxf(s, 1e-12f);
}

// ---------------- sparse update_features: per-(class, chunk) gather ----------------
// UFpart[z][k][d] = sum_{n in chunk z, label[n]==k%100} W4[n][k/100] * F[n][d]
__global__ __launch_bounds__(256)
void uf_class_part(const float* __restrict__ F, const float* __restrict__ W4,
                   const int* __restrict__ labels, float* __restrict__ UFpart) {
    __shared__ int lbl[CHUNK];    // 32 KB
    const int cls = blockIdx.x;
    long n0 = (long)blockIdx.y * CHUNK;
    int cnt = (int)(((long)NN - n0) < CHUNK ? (NN - n0) : CHUNK);
    const int tid = threadIdx.x;
    for (int i = tid; i < cnt; i += 256) lbl[i] = labels[n0 + i];
    __syncthreads();
    float acc[4][2] = {};
    for (int i = 0; i < cnt; ++i) {
        if (lbl[i] != cls) continue;                       // block-uniform branch
        long n = n0 + i;
        float w0 = W4[n * 4 + 0], w1 = W4[n * 4 + 1];
        float w2 = W4[n * 4 + 2], w3 = W4[n * 4 + 3];
        float f0 = F[n * DD + tid];
        float f1 = F[n * DD + tid + 256];
        acc[0][0] += w0 * f0; acc[0][1] += w0 * f1;
        acc[1][0] += w1 * f0; acc[1][1] += w1 * f1;
        acc[2][0] += w2 * f0; acc[2][1] += w2 * f1;
        acc[3][0] += w3 * f0; acc[3][1] += w3 * f1;
    }
#pragma unroll
    for (int p = 0; p < 4; ++p)
#pragma unroll
        for (int j = 0; j < 2; ++j)
            UFpart[((long)blockIdx.y * KK + cls + NCLS * p) * DD + tid + j * 256] = acc[p][j];
}

// ---------------- EMA prototype update + L2 normalize ----------------
__global__ __launch_bounds__(256)
void proto_update(const float* __restrict__ protos, const float* __restrict__ UFpart,
                  const float* __restrict__ umcol, __bf16* __restrict__ pnbf) {
    __shared__ float red[256];
    int k = blockIdx.x, tid = threadIdx.x;
    float inv_col = 1.0f / umcol[k];
    float v[2]; float ss = 0.f;
#pragma unroll
    for (int j = 0; j < 2; ++j) {
        int d = tid + j * 256;
        float uf = 0.f;
        for (int ch = 0; ch < NCHUNK; ++ch) uf += UFpart[((long)ch * KK + k) * DD + d];
        float x = 0.99f * protos[k * DD + d] + 0.01f * (uf * inv_col);
        v[j] = x; ss += x * x;
    }
    red[tid] = ss; __syncthreads();
    for (int s = 128; s; s >>= 1) { if (tid < s) red[tid] += red[tid + s]; __syncthreads(); }
    float nrm = fmaxf(sqrtf(red[0]), 1e-12f);
#pragma unroll
    for (int j = 0; j < 2; ++j) {
        int d = tid + j * 256;
        pnbf[k * DD + d] = (__bf16)(v[j] / nrm);
    }
}

// ---------------- prototype contrastive term ----------------
__global__ __launch_bounds__(128)
void proto_contra_rows(const float* __restrict__ G, float* __restrict__ rowvals) {
    __shared__ float red[128];
    int i = blockIdx.x, tid = threadIdx.x;
    float mx = -1e30f;
    for (int j = tid; j < KK; j += 128) mx = fmaxf(mx, 2.0f * G[i * KK + j]);
    red[tid] = mx; __syncthreads();
    for (int s = 64; s; s >>= 1) { if (tid < s) red[tid] = fmaxf(red[tid], red[tid + s]); __syncthreads(); }
    mx = red[0]; __syncthreads();
    float pos = 0.f, den = 0.f;
    int cls = i % NCLS;
    for (int j = tid; j < KK; j += 128) {
        float lg = 2.0f * G[i * KK + j] - mx;
        if (j != i) {
            den += __expf(lg);
            if (j % NCLS == cls) pos += lg;   // P-1 = 3 same-class off-diagonals
        }
    }
    red[tid] = den; __syncthreads();
    for (int s = 64; s; s >>= 1) { if (tid < s) red[tid] += red[tid + s]; __syncthreads(); }
    den = red[0]; __syncthreads();
    red[tid] = pos; __syncthreads();
    for (int s = 64; s; s >>= 1) { if (tid < s) red[tid] += red[tid + s]; __syncthreads(); }
    if (tid == 0) rowvals[i] = red[0] / 3.0f - __logf(den);
}

__global__ void reduce_psum(const float* __restrict__ rowvals, float* __restrict__ psum) {
    __shared__ float red[512];
    int tid = threadIdx.x;
    red[tid] = (tid < KK) ? rowvals[tid] : 0.f;
    __syncthreads();
    for (int s = 256; s; s >>= 1) { if (tid < s) red[tid] += red[tid + s]; __syncthreads(); }
    if (tid == 0) psum[0] = red[0];
}

// ---------------- last row pass of Sinkhorn #2 fused with per-sample loss ----------------
__global__ __launch_bounds__(256)
void fused_final(const float* __restrict__ S2, const float* __restrict__ r,
                 const int* __restrict__ labels, const float* __restrict__ psum,
                 float* __restrict__ out) {
    int wave = threadIdx.x >> 5, lane = threadIdx.x & 31;
    long n = (long)blockIdx.x * 8 + wave;
    if (n >= NN) return;
    float rsum = 0.f, se = 0.f;
    for (int k = lane; k < KK; k += 32) {
        float e20 = __expf(S2[n * KK + k] * INV_EPS);
        rsum += r[k] * e20;
        se += sqrtf(e20);                       // exp(10*s) = sqrt(exp(20*s))
    }
    for (int off = 16; off; off >>= 1) {
        rsum += __shfl_xor(rsum, off, 32);
        se   += __shfl_xor(se, off, 32);
    }
    if (lane == 0) {
        float neg = __logf(se);
        float cn = (float)NN / ((float)NN * fmaxf(rsum, 1e-12f));   // B * c[n]
        int l = labels[n];
        float q[4], a = 0.f;
#pragma unroll
        for (int p = 0; p < 4; ++p) {
            int k = l + NCLS * p;
            q[p] = r[k] * __expf(S2[n * KK + k] * INV_EPS) * cn;
            a += q[p];
        }
        a = fmaxf(a, 1e-12f);
        float pos = 0.f;
#pragma unroll
        for (int p = 0; p < 4; ++p) {
            int k = l + NCLS * p;
            pos += (q[p] / a) * (S2[n * KK + k] * INV_TEMP);
        }
        out[n] = -(pos - neg) - psum[0] / (float)KK;
    }
}

// ---------------- host launcher ----------------
extern "C" void kernel_launch(void* const* d_in, const int* in_sizes, int n_in,
                              void* d_out, int out_size, void* d_ws, size_t ws_size,
                              hipStream_t stream) {
    const float* features = (const float*)d_in[0];
    const float* protos   = (const float*)d_in[1];
    const int*   labels   = (const int*)d_in[2];
    float* out = (float*)d_out;

    char* ws = (char*)d_ws;
    size_t off = 0;
    auto alloc = [&](size_t bytes) -> void* {
        void* p = ws + off;
        off = (off + bytes + 255) & ~(size_t)255;
        return p;
    };
    float*  S      = (float*) alloc((size_t)NN * KK * 4);          // 160 MB (S1 then S2)
    __bf16* fbf    = (__bf16*)alloc((size_t)NN * DD * 2);          // 102.4 MB
    __bf16* pbf    = (__bf16*)alloc((size_t)KK * DD * 2);
    __bf16* pnbf   = (__bf16*)alloc((size_t)KK * DD * 2);
    float*  UFpart = (float*) alloc((size_t)NCHUNK * KK * DD * 4); // 10.65 MB
    float*  G      = (float*) alloc((size_t)KK * KK * 4);
    float*  rvec   = (float*) alloc((size_t)KK * 4);
    float*  colpart= (float*) alloc((size_t)NCB * KK * 4);         // 2.5 MB
    float*  W4     = (float*) alloc((size_t)NN * 4 * 4);
    float*  umpart = (float*) alloc((size_t)NUB * KK * 4);
    float*  umcol  = (float*) alloc((size_t)KK * 4);
    float*  rowvals= (float*) alloc((size_t)KK * 4);
    float*  psum   = (float*) alloc(256);

    // 1. bf16 operand conversion
    conv_bf16<<<((long)NN * DD + 255) / 256, 256, 0, stream>>>(features, fbf, (long)NN * DD);
    conv_bf16<<<(KK * DD + 255) / 256, 256, 0, stream>>>(protos, pbf, (long)KK * DD);

    // 2. S1 = F * P^T
    gemm_nt<<<dim3((NN + 127) / 128, (KK + 63) / 64), 256, 0, stream>>>(fbf, pbf, S, NN, KK);

    // 3. Sinkhorn #1 (4 streaming passes over S)
    sink_colpass1<<<NCB, 512, 0, stream>>>(S, colpart);
    sink_colreduce<<<1, 512, 0, stream>>>(colpart, rvec);
    sink_fused<<<NCB, 256, 0, stream>>>(S, rvec, colpart);
    sink_colreduce<<<1, 512, 0, stream>>>(colpart, rvec);
    sink_fused<<<NCB, 256, 0, stream>>>(S, rvec, colpart);
    sink_colreduce<<<1, 512, 0, stream>>>(colpart, rvec);
    fused_w4<<<NN / 8, 256, 0, stream>>>(S, rvec, labels, W4);

    // 4. update_mask column sums
    um_colpart<<<NUB, 512, 0, stream>>>(W4, labels, umpart);
    um_colreduce<<<1, 512, 0, stream>>>(umpart, umcol);

    // 5. update_features via label-sparsity (fp32, deterministic split-N partials)
    uf_class_part<<<dim3(NCLS, NCHUNK), 256, 0, stream>>>(features, W4, labels, UFpart);

    // 6. protos_new = l2(0.99*P + 0.01*UF)
    proto_update<<<KK, 256, 0, stream>>>(protos, UFpart, umcol, pnbf);

    // 7. S2 = F * Pnew^T (reuse S)
    gemm_nt<<<dim3((NN + 127) / 128, (KK + 63) / 64), 256, 0, stream>>>(fbf, pnbf, S, NN, KK);

    // 8. proto-contra scalar (needed by fused_final)
    gemm_nt<<<dim3((KK + 127) / 128, (KK + 63) / 64), 256, 0, stream>>>(pnbf, pnbf, G, KK, KK);
    proto_contra_rows<<<KK, 128, 0, stream>>>(G, rowvals);
    reduce_psum<<<1, 512, 0, stream>>>(rowvals, psum);

    // 9. Sinkhorn #2 + fused loss epilogue
    sink_colpass1<<<NCB, 512, 0, stream>>>(S, colpart);
    sink_colreduce<<<1, 512, 0, stream>>>(colpart, rvec);
    sink_fused<<<NCB, 256, 0, stream>>>(S, rvec, colpart);
    sink_colreduce<<<1, 512, 0, stream>>>(colpart, rvec);
    sink_fused<<<NCB, 256, 0, stream>>>(S, rvec, colpart);
    sink_colreduce<<<1, 512, 0, stream>>>(colpart, rvec);
    fused_final<<<NN / 8, 256, 0, stream>>>(S, rvec, labels, psum, out);
}
